// VideoPrismEncoderBlock_33328946217696
// MI455X (gfx1250) — matrix operand order repaired and checked by reading, not compiled, not run on hardware
//
#include <hip/hip_runtime.h>

// ---------------------------------------------------------------------------
// CDNA5 (gfx1250) VideoPrism encoder block: LN -> attention (softcap) -> LN ->
// MLP(GELU). All GEMMs in bf16 via v_wmma_f32_16x16x32_bf16, f32 accumulate.
// Global->LDS staging via global_load_async_to_lds_b128 (ASYNCcnt) where the
// copy is layout-preserving; double-buffered LDS in the GEMM.
// ---------------------------------------------------------------------------

typedef __attribute__((ext_vector_type(16))) __bf16 v16bf;
typedef __attribute__((ext_vector_type(8)))  __bf16 v8bf;
typedef __attribute__((ext_vector_type(8)))  float  v8f;

union V16 { v16bf v; v8bf h[2]; };

__device__ __forceinline__ unsigned short f2bf(float x) {
    unsigned int u = __float_as_uint(x);
    u += 0x7FFFu + ((u >> 16) & 1u);   // round-to-nearest-even
    return (unsigned short)(u >> 16);
}

__device__ __forceinline__ void async_cp16(unsigned lds_addr, unsigned gbyteoff,
                                           const void* base) {
    asm volatile("global_load_async_to_lds_b128 %0, %1, %2"
                 :: "v"(lds_addr), "v"(gbyteoff), "s"((unsigned long long)base)
                 : "memory");
}
__device__ __forceinline__ void async_wait0() {
    asm volatile("s_wait_asynccnt 0x0" ::: "memory");
}

// ---------------------------------------------------------------------------
// Weight transpose + f32->bf16 convert:  out[C][R] = in[R][C] * scale
// ---------------------------------------------------------------------------
__global__ void tconv_kernel(const float* __restrict__ in,
                             unsigned short* __restrict__ out,
                             int R, int C, float scale) {
    size_t idx = (size_t)blockIdx.x * 256 + threadIdx.x;
    if (idx >= (size_t)R * C) return;
    int i = (int)(idx / C), j = (int)(idx % C);
    out[(size_t)j * R + i] = f2bf(in[idx] * scale);
}

__global__ void qkvbias_kernel(const float* __restrict__ bq,
                               const float* __restrict__ bk,
                               const float* __restrict__ bv,
                               float* __restrict__ out) {
    int i = blockIdx.x * 256 + threadIdx.x;
    if (i < 1024)      out[i] = bq[i] * 0.125f;     // fold 1/sqrt(64) into q
    else if (i < 2048) out[i] = bk[i - 1024];
    else if (i < 3072) out[i] = bv[i - 2048];
}

// ---------------------------------------------------------------------------
// LayerNorm: one block per row of 1024, output bf16
// ---------------------------------------------------------------------------
__global__ __launch_bounds__(256)
void ln_kernel(const float* __restrict__ x, const float* __restrict__ gamma,
               const float* __restrict__ beta, unsigned short* __restrict__ out) {
    __shared__ float red[256];
    const int tid = threadIdx.x;
    const float* xr = x + (size_t)blockIdx.x * 1024;
    float v[4]; float s = 0.f;
    for (int i = 0; i < 4; ++i) { v[i] = xr[tid + i * 256]; s += v[i]; }
    red[tid] = s; __syncthreads();
    for (int o = 128; o > 0; o >>= 1) { if (tid < o) red[tid] += red[tid + o]; __syncthreads(); }
    float mean = red[0] * (1.f / 1024.f); __syncthreads();
    float s2 = 0.f;
    for (int i = 0; i < 4; ++i) { float d = v[i] - mean; s2 += d * d; }
    red[tid] = s2; __syncthreads();
    for (int o = 128; o > 0; o >>= 1) { if (tid < o) red[tid] += red[tid + o]; __syncthreads(); }
    float rstd = rsqrtf(red[0] * (1.f / 1024.f) + 1e-6f);
    unsigned short* orow = out + (size_t)blockIdx.x * 1024;
    for (int i = 0; i < 4; ++i) {
        int c = tid + i * 256;
        orow[c] = f2bf((v[i] - mean) * rstd * gamma[c] + beta[c]);
    }
}

// ---------------------------------------------------------------------------
// Tiled GEMM: C[M,N] = A[M,K](bf16,row) * Wt[N,K](bf16,row)^T + bias
//   EP=0: bf16 out; EP=1: bf16 GELU(exact) out; EP=2: f32 out + residual
// Block 256 thr (8 waves), tile 128x128, K-step 32; wave tile 32x64 (2x4 WMMA).
// Double-buffered LDS staged with global_load_async_to_lds_b128 (ASYNCcnt).
// ---------------------------------------------------------------------------
template <int EP>
__global__ __launch_bounds__(256)
void gemm_kernel(const unsigned short* __restrict__ A,
                 const unsigned short* __restrict__ Wt,
                 const float* __restrict__ bias,
                 const float* __restrict__ resid,
                 void* __restrict__ outp, int M, int N, int K) {
    __shared__ alignas(16) unsigned short As[2][128 * 40];
    __shared__ alignas(16) unsigned short Bs[2][128 * 40];
    const int t = threadIdx.x;
    const int w = t >> 5, lane = t & 31;
    const int lhi = lane >> 4, lm = lane & 15;
    const int m0 = blockIdx.y * 128, n0 = blockIdx.x * 128;
    const int wrow = (w & 3) * 32;   // wave row base: 0/32/64/96
    const int wcol = (w >> 2) * 64;  // wave col base: 0/64

    // async stage of one 128x32 A tile + 128x32 B tile into buffer `buf`
    auto stage_async = [&](int buf, int k0) {
        for (int c = 0; c < 2; ++c) {
            int idx = t * 2 + c;
            int r = idx >> 2, kc = (idx & 3) * 8;          // r:0..127, kc:0..24
            async_cp16((unsigned)(uintptr_t)&As[buf][r * 40 + kc],
                       (unsigned)((((size_t)(m0 + r)) * K + k0 + kc) * 2), A);
            async_cp16((unsigned)(uintptr_t)&Bs[buf][r * 40 + kc],
                       (unsigned)((((size_t)(n0 + r)) * K + k0 + kc) * 2), Wt);
        }
    };

    v8f acc[2][4] = {};
    const int nk = K >> 5;

    stage_async(0, 0);
    async_wait0();
    __syncthreads();

    for (int kb = 0; kb < nk; ++kb) {
        const int cur = kb & 1;
        if (kb + 1 < nk) stage_async(cur ^ 1, (kb + 1) * 32);

        V16 a[2], b[4];
        for (int i = 0; i < 2; ++i) {
            int m = wrow + i * 16 + lm;
            a[i].h[0] = *(const v8bf*)(&As[cur][m * 40 + 8 * lhi]);       // k=8hi..
            a[i].h[1] = *(const v8bf*)(&As[cur][m * 40 + 16 + 8 * lhi]);  // k=16+8hi..
        }
        for (int j = 0; j < 4; ++j) {
            int n = wcol + j * 16 + lm;
            b[j].h[0] = *(const v8bf*)(&Bs[cur][n * 40 + 8 * lhi]);       // k=8hi..
            b[j].h[1] = *(const v8bf*)(&Bs[cur][n * 40 + 16 + 8 * lhi]);
        }
        for (int i = 0; i < 2; ++i)
            for (int j = 0; j < 4; ++j)
                acc[i][j] = __builtin_amdgcn_wmma_f32_16x16x32_bf16(
                    false, a[i].v, false, b[j].v, (short)0, acc[i][j], false, false);

        async_wait0();
        __syncthreads();
    }

    for (int i = 0; i < 2; ++i)
        for (int j = 0; j < 4; ++j) {
            int gn = n0 + wcol + j * 16 + lm;
            float bv = bias[gn];
            for (int r = 0; r < 8; ++r) {
                int gm = m0 + wrow + i * 16 + r + 8 * lhi;
                float v = acc[i][j][r] + bv;
                size_t off = (size_t)gm * N + gn;
                if (EP == 1) v = 0.5f * v * (1.0f + erff(v * 0.70710678118f));
                if (EP == 2) ((float*)outp)[off] = v + resid[off];
                else         ((unsigned short*)outp)[off] = f2bf(v);
            }
        }
}

// ---------------------------------------------------------------------------
// Flash attention w/ logit soft cap. One block = (b, head, 64 q rows).
// Key chunks of 64; online softmax (4-way parallel per row); WMMA bf16.
// qkv layout: [b*T + t][3072], q at head*64, k at +1024, v at +2048.
// ---------------------------------------------------------------------------
__global__ __launch_bounds__(256)
void attn_kernel(const unsigned short* __restrict__ qkv,
                 unsigned short* __restrict__ ctx) {
    __shared__ alignas(16) unsigned short Qs[64 * 72];
    __shared__ alignas(16) unsigned short Ks[64 * 72];
    __shared__ alignas(16) unsigned short Vt[64 * 72];   // [h][s]
    __shared__ alignas(16) unsigned short Pb[64 * 72];
    __shared__ alignas(16) float Sb[64 * 68];
    __shared__ float rowM[64], rowL[64], rowScale[64];
    __shared__ float redmax[4][64], redsum[4][64];

    const int t = threadIdx.x, w = t >> 5, lane = t & 31;
    const int lhi = lane >> 4, lm = lane & 15;
    const int bx = blockIdx.x;
    const int qt = bx & 31, head = (bx >> 5) & 15, b = bx >> 9;
    const size_t base = ((size_t)b * 2048) * 3072 + head * 64;
    const int q0 = qt * 64;
    const int mt = w & 3;          // wave's 16-row q tile
    const int ct0 = (w >> 2) * 2;  // wave's pair of 16-col tiles
    const int srow = t & 63;       // softmax: row owned
    const int sg = t >> 6;         // softmax: column group 0..3

    for (int c = 0; c < 2; ++c) {  // stage Q 64x64 (async to LDS)
        int idx = t * 2 + c, r = idx >> 3, kc = (idx & 7) * 8;
        async_cp16((unsigned)(uintptr_t)&Qs[r * 72 + kc],
                   (unsigned)((base + (size_t)(q0 + r) * 3072 + kc) * 2), qkv);
    }
    if (t < 64) { rowM[t] = -1e30f; rowL[t] = 0.f; }
    v8f accO[2] = {};
    async_wait0();
    __syncthreads();

    for (int scn = 0; scn < 32; ++scn) {
        const int s0 = scn * 64;
        for (int c = 0; c < 2; ++c) {  // K async row-major; V via VGPR transpose
            int idx = t * 2 + c, r = idx >> 3, kc = (idx & 7) * 8;
            async_cp16((unsigned)(uintptr_t)&Ks[r * 72 + kc],
                       (unsigned)((base + 1024 + (size_t)(s0 + r) * 3072 + kc) * 2), qkv);
            uint4 vv = *(const uint4*)(qkv + base + 2048 + (size_t)(s0 + r) * 3072 + kc);
            const unsigned short* hv = (const unsigned short*)&vv;
            for (int e = 0; e < 8; ++e) Vt[(kc + e) * 72 + r] = hv[e];
        }
        async_wait0();
        __syncthreads();

        // S = Q * K^T (16x16 tiles), softcap, spill to LDS
        for (int j = 0; j < 2; ++j) {
            v8f s = {};
            int qm = mt * 16 + lm;
            int sn = (ct0 + j) * 16 + lm;
            for (int ks = 0; ks < 64; ks += 32) {
                V16 a, bb;
                a.h[0]  = *(const v8bf*)(Qs + qm * 72 + ks + 8 * lhi);
                a.h[1]  = *(const v8bf*)(Qs + qm * 72 + ks + 16 + 8 * lhi);
                bb.h[0] = *(const v8bf*)(Ks + sn * 72 + ks + 8 * lhi);
                bb.h[1] = *(const v8bf*)(Ks + sn * 72 + ks + 16 + 8 * lhi);
                s = __builtin_amdgcn_wmma_f32_16x16x32_bf16(
                    false, a.v, false, bb.v, (short)0, s, false, false);
            }
            for (int r = 0; r < 8; ++r) {
                int row = mt * 16 + r + 8 * lhi;
                Sb[row * 68 + (ct0 + j) * 16 + lm] = 50.f * tanhf(s[r] * 0.02f);
            }
        }
        __syncthreads();

        // online softmax, 4 column-groups of 16 per row
        {
            float cm = -1e30f;
            for (int c2 = 0; c2 < 16; ++c2)
                cm = fmaxf(cm, Sb[srow * 68 + sg * 16 + c2]);
            redmax[sg][srow] = cm;
        }
        __syncthreads();
        {
            float cm = fmaxf(fmaxf(redmax[0][srow], redmax[1][srow]),
                             fmaxf(redmax[2][srow], redmax[3][srow]));
            float nm = fmaxf(rowM[srow], cm);
            float sum = 0.f;
            for (int c2 = 0; c2 < 16; ++c2) {
                float p = __expf(Sb[srow * 68 + sg * 16 + c2] - nm);
                Pb[srow * 72 + sg * 16 + c2] = f2bf(p);
                sum += p;
            }
            redsum[sg][srow] = sum;
        }
        __syncthreads();
        if (t < 64) {
            float cm = fmaxf(fmaxf(redmax[0][t], redmax[1][t]),
                             fmaxf(redmax[2][t], redmax[3][t]));
            float nm = fmaxf(rowM[t], cm);
            float corr = __expf(rowM[t] - nm);
            rowL[t] = rowL[t] * corr +
                      (redsum[0][t] + redsum[1][t] + redsum[2][t] + redsum[3][t]);
            rowScale[t] = corr;
            rowM[t] = nm;
        }
        __syncthreads();

        // O = O*corr + P*V
        float sc8[8];
        for (int r = 0; r < 8; ++r) sc8[r] = rowScale[mt * 16 + r + 8 * lhi];
        for (int j = 0; j < 2; ++j)
            for (int r = 0; r < 8; ++r) accO[j][r] *= sc8[r];
        for (int j = 0; j < 2; ++j) {
            int qm = mt * 16 + lm;
            int hn = (ct0 + j) * 16 + lm;
            for (int ks = 0; ks < 64; ks += 32) {
                V16 a, bb;
                a.h[0]  = *(const v8bf*)(Pb + qm * 72 + ks + 8 * lhi);
                a.h[1]  = *(const v8bf*)(Pb + qm * 72 + ks + 16 + 8 * lhi);
                bb.h[0] = *(const v8bf*)(Vt + hn * 72 + ks + 8 * lhi);
                bb.h[1] = *(const v8bf*)(Vt + hn * 72 + ks + 16 + 8 * lhi);
                accO[j] = __builtin_amdgcn_wmma_f32_16x16x32_bf16(
                    false, a.v, false, bb.v, (short)0, accO[j], false, false);
            }
        }
        __syncthreads();
    }

    float inv[8];
    for (int r = 0; r < 8; ++r) inv[r] = 1.0f / rowL[mt * 16 + r + 8 * lhi];
    for (int j = 0; j < 2; ++j) {
        int hcol = head * 64 + (ct0 + j) * 16 + lm;
        for (int r = 0; r < 8; ++r) {
            int row = q0 + mt * 16 + r + 8 * lhi;
            ctx[((size_t)b * 2048 + row) * 1024 + hcol] = f2bf(accO[j][r] * inv[r]);
        }
    }
}

// ---------------------------------------------------------------------------
extern "C" void kernel_launch(void* const* d_in, const int* in_sizes, int n_in,
                              void* d_out, int out_size, void* d_ws, size_t ws_size,
                              hipStream_t stream) {
    const float* inputs    = (const float*)d_in[0];
    const float* ln1_gamma = (const float*)d_in[1];
    const float* ln1_beta  = (const float*)d_in[2];
    const float* wq        = (const float*)d_in[3];
    const float* bq        = (const float*)d_in[4];
    const float* wk        = (const float*)d_in[5];
    const float* bk        = (const float*)d_in[6];
    const float* wv        = (const float*)d_in[7];
    const float* bv        = (const float*)d_in[8];
    const float* wo        = (const float*)d_in[9];
    const float* bo        = (const float*)d_in[10];
    const float* ln2_gamma = (const float*)d_in[11];
    const float* ln2_beta  = (const float*)d_in[12];
    const float* w1        = (const float*)d_in[13];
    const float* b1        = (const float*)d_in[14];
    const float* w2        = (const float*)d_in[15];
    const float* b2        = (const float*)d_in[16];

    const int M = 8192;  // B*T

    char* p = (char*)d_ws;
    auto alloc = [&](size_t bytes) {
        char* r = p;
        p += (bytes + 255) & ~(size_t)255;
        return r;
    };
    unsigned short* xn     = (unsigned short*)alloc((size_t)M * 1024 * 2);
    unsigned short* wqkv_t = (unsigned short*)alloc((size_t)3072 * 1024 * 2);
    float*          qb     = (float*)alloc(3072 * 4);
    unsigned short* qkv    = (unsigned short*)alloc((size_t)M * 3072 * 2);
    unsigned short* wot    = (unsigned short*)alloc((size_t)1024 * 1024 * 2);
    unsigned short* cx     = (unsigned short*)alloc((size_t)M * 1024 * 2);
    float*          xres   = (float*)alloc((size_t)M * 1024 * 4);
    unsigned short* ybf    = (unsigned short*)alloc((size_t)M * 1024 * 2);
    unsigned short* w1t    = (unsigned short*)alloc((size_t)4096 * 1024 * 2);
    unsigned short* hbf    = (unsigned short*)alloc((size_t)M * 4096 * 2);
    unsigned short* w2t    = (unsigned short*)alloc((size_t)1024 * 4096 * 2);

    // weight prep (bf16, [N][K] transposed; q scale folded)
    int g1m = (1024 * 1024 + 255) / 256;
    int g4m = (1024 * 4096 + 255) / 256;
    tconv_kernel<<<g1m, 256, 0, stream>>>(wq, wqkv_t,                 1024, 1024, 0.125f);
    tconv_kernel<<<g1m, 256, 0, stream>>>(wk, wqkv_t + 1024 * 1024,   1024, 1024, 1.0f);
    tconv_kernel<<<g1m, 256, 0, stream>>>(wv, wqkv_t + 2048 * 1024,   1024, 1024, 1.0f);
    tconv_kernel<<<g1m, 256, 0, stream>>>(wo, wot,                    1024, 1024, 1.0f);
    tconv_kernel<<<g4m, 256, 0, stream>>>(w1, w1t,                    1024, 4096, 1.0f);
    tconv_kernel<<<g4m, 256, 0, stream>>>(w2, w2t,                    4096, 1024, 1.0f);
    qkvbias_kernel<<<12, 256, 0, stream>>>(bq, bk, bv, qb);

    // LN1 -> QKV -> attention -> out-proj(+residual)
    ln_kernel<<<M, 256, 0, stream>>>(inputs, ln1_gamma, ln1_beta, xn);
    gemm_kernel<0><<<dim3(3072 / 128, M / 128), 256, 0, stream>>>(
        xn, wqkv_t, qb, nullptr, qkv, M, 3072, 1024);
    attn_kernel<<<2048, 256, 0, stream>>>(qkv, cx);
    gemm_kernel<2><<<dim3(1024 / 128, M / 128), 256, 0, stream>>>(
        cx, wot, bo, inputs, xres, M, 1024, 1024);

    // LN2 -> MLP1(GELU) -> MLP2(+residual) -> out
    ln_kernel<<<M, 256, 0, stream>>>(xres, ln2_gamma, ln2_beta, ybf);
    gemm_kernel<1><<<dim3(4096 / 128, M / 128), 256, 0, stream>>>(
        ybf, w1t, b1, nullptr, hbf, M, 4096, 1024);
    gemm_kernel<2><<<dim3(1024 / 128, M / 128), 256, 0, stream>>>(
        hbf, w2t, b2, xres, (float*)d_out, M, 1024, 4096);
}
